// NASNetwork_51092930953536
// MI455X (gfx1250) — matrix operand order repaired and checked by reading, not compiled, not run on hardware
//
#include <hip/hip_runtime.h>

#define HID 2048
#define GATES 8192
#define NNODES 7
#define NOPS 11
#define TEMPR 5.0f
#define TANHC 2.5f

typedef __attribute__((ext_vector_type(16))) __bf16 v16bf;
typedef __attribute__((ext_vector_type(8)))  __bf16 v8bf;
typedef __attribute__((ext_vector_type(8)))  float  v8f;

struct NasState {
    int inp_idx;
    unsigned key0, key1;
    float log_p, entropy;
};

__device__ __forceinline__ __bf16 f32_to_bf16(float f) {
    unsigned u = __builtin_bit_cast(unsigned, f);
    unsigned r;
    if ((u & 0x7F800000u) == 0x7F800000u) r = u;              // Inf/NaN: truncate
    else r = u + 0x7FFFu + ((u >> 16) & 1u);                   // round-to-nearest-even
    return __builtin_bit_cast(__bf16, (unsigned short)(r >> 16));
}

// ---------------------------------------------------------------------------
// Setup kernels (run every launch; d_ws is not preserved between calls)
// ---------------------------------------------------------------------------

// Convert a rows x HID f32 matrix into bf16 tiled WMMA-B layout:
//   dst[((nt*64 + kt)*32 + lane)*16 + i] = src[nt*16 + (lane&15)][kt*32 + (lane>>4)*8 + (i&7) + (i>>3)*16]
__global__ void k_convert_swizzle(const float* __restrict__ src,
                                  __bf16* __restrict__ dst, long total) {
    long o = (long)blockIdx.x * 256 + threadIdx.x;
    if (o >= total) return;
    int  i    = (int)(o & 15);
    int  lane = (int)((o >> 4) & 31);
    long tile = o >> 9;
    int  kt   = (int)(tile % (HID / 32));
    long nt   = tile / (HID / 32);
    long row  = nt * 16 + (lane & 15);
    int  col  = kt * 32 + (lane >> 4) * 8 + (i & 7) + (i >> 3) * 16;
    dst[o] = f32_to_bf16(src[row * HID + col]);
}

// Plain elementwise f32 -> bf16 (for emb rows used as WMMA A operand)
__global__ void k_convert_row(const float* __restrict__ src,
                              __bf16* __restrict__ dst, long total) {
    long o = (long)blockIdx.x * 256 + threadIdx.x;
    if (o < total) dst[o] = f32_to_bf16(src[o]);
}

// P[r][n] = dot(emb[r], W_emb_attn[n]) for r<8  (loop-invariant attention term)
__global__ void k_precompute_P(const float* __restrict__ emb,
                               const float* __restrict__ Wemb,
                               float* __restrict__ P) {
    int gid = blockIdx.x * 256 + threadIdx.x;      // 16384 total
    int r = gid >> 11, n = gid & (HID - 1);
    const float* e = emb + (long)r * HID;
    const float* w = Wemb + (long)n * HID;
    float s = 0.f;
    for (int k = 0; k < HID; ++k) s += e[k] * w[k];
    P[gid] = s;
}

__global__ void k_init(__bf16* h_bf, __bf16* zbuf, float* h_f, float* c,
                       NasState* st) {
    const __bf16 z = __builtin_bit_cast(__bf16, (unsigned short)0);
    for (int i = threadIdx.x; i < HID; i += 256) {
        h_bf[i] = z;
        zbuf[i] = z;
        h_f[i]  = 0.f;
        c[i]    = 0.f;
    }
    if (threadIdx.x == 0) {
        st->inp_idx = -1;          // step 0 input embedding is zeros
        st->key0 = 0u;             // jax.random.key(42) -> (0, 42)
        st->key1 = 42u;
        st->log_p = 0.f;
        st->entropy = 0.f;
    }
}

// ---------------------------------------------------------------------------
// WMMA matvec kernels (bandwidth-bound on weight streaming from L2)
//
// Split-K x4: all 4 waves of a block share one 16-output n-tile, each covering
// 512 K-elements (16 x K32 WMMA steps), partials reduced through LDS.
// A-operand loads are unconditional: non-loader lanes read a zero buffer,
// eliminating per-iteration EXEC juggling. B tiles are double-buffered so
// WMMA overlaps the next tile's loads.
// ---------------------------------------------------------------------------

#define SHUF16(lo, hi) __builtin_shufflevector(lo, hi, 0,1,2,3,4,5,6,7,8,9,10,11,12,13,14,15)

// gates[n] = x.W_ih[n] + h.W_hh[n] + b_ih[n] + b_hh[n];  grid = 512 blocks x 128
__global__ void __launch_bounds__(128) k_gates(
    const __bf16* __restrict__ Wih, const __bf16* __restrict__ Whh,
    const __bf16* __restrict__ emb_bf, const __bf16* __restrict__ h_bf,
    const __bf16* __restrict__ zbuf,
    const float* __restrict__ b_ih, const float* __restrict__ b_hh,
    const int* __restrict__ inp_ptr, float* __restrict__ gates) {
    __shared__ float part[4][16];
    int wave = threadIdx.x >> 5, lane = threadIdx.x & 31;
    int n_tile = blockIdx.x;
    int inp = *inp_ptr;
    bool loaderA = ((lane & 15) == 0);
    int kh = lane >> 4;

    const __bf16* xrow = (inp >= 0) ? (emb_bf + (long)inp * HID) : zbuf;
    const __bf16* xb = loaderA ? xrow : zbuf;
    const __bf16* hb = loaderA ? h_bf : zbuf;

    int kt0 = wave * 16;                                   // this wave's K chunk
    const __bf16* bip = Wih + ((long)n_tile * 64 + kt0) * 512 + lane * 16;
    const __bf16* bhp = Whh + ((long)n_tile * 64 + kt0) * 512 + lane * 16;

    v8f acc = {};
    v16bf bi = *(const v16bf*)bip;
    v16bf bh = *(const v16bf*)bhp;
#pragma unroll
    for (int j = 0; j < 16; ++j) {
        int base = (kt0 + j) * 32 + kh * 8;
        v8bf xlo = *(const v8bf*)(xb + base);
        v8bf xhi = *(const v8bf*)(xb + base + 16);
        v8bf hlo = *(const v8bf*)(hb + base);
        v8bf hhi = *(const v8bf*)(hb + base + 16);
        v16bf bi_n = bi, bh_n = bh;
        if (j < 15) {                                      // prefetch next B tiles
            bi_n = *(const v16bf*)(bip + (j + 1) * 512);
            bh_n = *(const v16bf*)(bhp + (j + 1) * 512);
        }
        v16bf ax = SHUF16(xlo, xhi);
        v16bf ah = SHUF16(hlo, hhi);
        acc = __builtin_amdgcn_wmma_f32_16x16x32_bf16(false, ax, false, bi,
                                                      (short)0, acc, false, false);
        acc = __builtin_amdgcn_wmma_f32_16x16x32_bf16(false, ah, false, bh,
                                                      (short)0, acc, false, false);
        bi = bi_n; bh = bh_n;
    }
    if (lane < 16) part[wave][lane] = acc[0];              // D row M=0
    __syncthreads();
    if (threadIdx.x < 16) {
        int n = n_tile * 16 + threadIdx.x;
        float s = part[0][threadIdx.x] + part[1][threadIdx.x] +
                  part[2][threadIdx.x] + part[3][threadIdx.x];
        gates[n] = s + b_ih[n] + b_hh[n];
    }
}

// out[n] = h . W[n]   (a_hid = h @ W_hid_attn^T);  grid = 128 blocks x 128
__global__ void __launch_bounds__(128) k_ahid(
    const __bf16* __restrict__ W, const __bf16* __restrict__ h_bf,
    const __bf16* __restrict__ zbuf, float* __restrict__ out) {
    __shared__ float part[4][16];
    int wave = threadIdx.x >> 5, lane = threadIdx.x & 31;
    int n_tile = blockIdx.x;
    bool loaderA = ((lane & 15) == 0);
    int kh = lane >> 4;
    const __bf16* hb = loaderA ? h_bf : zbuf;

    int kt0 = wave * 16;
    const __bf16* bp = W + ((long)n_tile * 64 + kt0) * 512 + lane * 16;

    v8f acc = {};
    v16bf b = *(const v16bf*)bp;
#pragma unroll
    for (int j = 0; j < 16; ++j) {
        int base = (kt0 + j) * 32 + kh * 8;
        v8bf hlo = *(const v8bf*)(hb + base);
        v8bf hhi = *(const v8bf*)(hb + base + 16);
        v16bf b_n = b;
        if (j < 15) b_n = *(const v16bf*)(bp + (j + 1) * 512);
        v16bf ah = SHUF16(hlo, hhi);
        acc = __builtin_amdgcn_wmma_f32_16x16x32_bf16(false, ah, false, b,
                                                      (short)0, acc, false, false);
        b = b_n;
    }
    if (lane < 16) part[wave][lane] = acc[0];
    __syncthreads();
    if (threadIdx.x < 16) {
        out[n_tile * 16 + threadIdx.x] =
            part[0][threadIdx.x] + part[1][threadIdx.x] +
            part[2][threadIdx.x] + part[3][threadIdx.x];
    }
}

// ---------------------------------------------------------------------------
// LSTM pointwise
// ---------------------------------------------------------------------------
__global__ void k_lstm(const float* __restrict__ g, float* __restrict__ c,
                       float* __restrict__ h_f, __bf16* __restrict__ h_bf) {
    int idx = blockIdx.x * 256 + threadIdx.x;     // 2048 total
    float gi = g[idx], gf = g[HID + idx], gg = g[2 * HID + idx], go = g[3 * HID + idx];
    float si = 1.f / (1.f + expf(-gi));
    float sf = 1.f / (1.f + expf(-gf));
    float so = 1.f / (1.f + expf(-go));
    float c2 = sf * c[idx] + si * tanhf(gg);
    float h2 = so * tanhf(c2);
    c[idx] = c2;
    h_f[idx] = h2;
    h_bf[idx] = f32_to_bf16(h2);
}

// ---------------------------------------------------------------------------
// Threefry2x32 (JAX-compatible) + categorical sampling
// ---------------------------------------------------------------------------
__device__ __forceinline__ void threefry2x32(unsigned k0, unsigned k1,
                                             unsigned c0, unsigned c1,
                                             unsigned& o0, unsigned& o1) {
    unsigned ks0 = k0, ks1 = k1, ks2 = k0 ^ k1 ^ 0x1BD11BDAu;
    unsigned x0 = c0 + ks0, x1 = c1 + ks1;
#define TF_RND(r) { x0 += x1; x1 = (x1 << (r)) | (x1 >> (32 - (r))); x1 ^= x0; }
    TF_RND(13) TF_RND(15) TF_RND(26) TF_RND(6)
    x0 += ks1; x1 += ks2 + 1u;
    TF_RND(17) TF_RND(29) TF_RND(16) TF_RND(24)
    x0 += ks2; x1 += ks0 + 2u;
    TF_RND(13) TF_RND(15) TF_RND(26) TF_RND(6)
    x0 += ks0; x1 += ks1 + 3u;
    TF_RND(17) TF_RND(29) TF_RND(16) TF_RND(24)
    x0 += ks1; x1 += ks2 + 4u;
    TF_RND(13) TF_RND(15) TF_RND(26) TF_RND(6)
    x0 += ks2; x1 += ks0 + 5u;
#undef TF_RND
    o0 = x0; o1 = x1;
}

__device__ void sample_common(const float* lg, int n, NasState* st,
                              float* out, int slot, bool is_node) {
    // skey, sk = jax.random.split(skey): threefry over iota(4), pairs (0,2),(1,3)
    unsigned a0, a1, b0, b1;
    threefry2x32(st->key0, st->key1, 0u, 2u, a0, a1);
    threefry2x32(st->key0, st->key1, 1u, 3u, b0, b1);
    st->key0 = a0; st->key1 = b0;                 // new skey
    unsigned sk0 = a1, sk1 = b1;                  // sampling key

    // random_bits(sk, 32, (n,)): iota(n) (zero-padded if odd), pairs (p, m+p)
    unsigned bits[12];
    int m = (n + 1) / 2;
    for (int p = 0; p < m; ++p) {
        unsigned c0 = (unsigned)p;
        unsigned c1 = (m + p < n) ? (unsigned)(m + p) : 0u;
        unsigned o0, o1;
        threefry2x32(sk0, sk1, c0, c1, o0, o1);
        bits[p] = o0;
        if (m + p < n) bits[m + p] = o1;
    }

    // gumbel argmax (jax.random.categorical)
    const float tiny = 1.1754943508222875e-38f;
    int action = 0; float best = -1e30f;
    for (int i = 0; i < n; ++i) {
        float f = __builtin_bit_cast(float, 0x3F800000u | (bits[i] >> 9)) - 1.0f;
        float u = f * (1.0f - tiny) + tiny;
        if (u < tiny) u = tiny;
        float gmb = -logf(-logf(u));
        float v = lg[i] + gmb;
        if (v > best) { best = v; action = i; }
    }

    // log-softmax, entropy, log_p
    float lmax = lg[0];
    for (int i = 1; i < n; ++i) lmax = fmaxf(lmax, lg[i]);
    float s = 0.f;
    for (int i = 0; i < n; ++i) s += expf(lg[i] - lmax);
    float lse = logf(s);
    float ent = 0.f;
    for (int i = 0; i < n; ++i) {
        float lp = lg[i] - lmax - lse;
        ent -= lp * expf(lp);
    }
    st->log_p   += lg[action] - lmax - lse;
    st->entropy += ent;
    st->inp_idx  = is_node ? action : (action + NNODES + 1);

    out[slot] = (float)action;
    out[28] = st->log_p;
    out[29] = st->entropy;
}

// logits[r] = TANHC * tanh( (W_v . tanh(P[r] + a_hid)) / TEMP ), r < n
__global__ void __launch_bounds__(256) k_sample_node(
    const float* __restrict__ P, const float* __restrict__ a_hid,
    const float* __restrict__ Wv, int n, NasState* st, float* out, int slot) {
    __shared__ float red[256];
    __shared__ float lg[8];
    int tid = threadIdx.x;
    for (int r = 0; r < n; ++r) {
        float p = 0.f;
        for (int k = tid; k < HID; k += 256)
            p += tanhf(P[r * HID + k] + a_hid[k]) * Wv[k];
        red[tid] = p; __syncthreads();
        for (int s = 128; s > 0; s >>= 1) {
            if (tid < s) red[tid] += red[tid + s];
            __syncthreads();
        }
        if (tid == 0) lg[r] = TANHC * tanhf(red[0] / TEMPR);
        __syncthreads();
    }
    if (tid == 0) sample_common(lg, n, st, out, slot, true);
}

// logits[j] = (TANHC/OP_TANH_RED) * tanh( (h.W_soft[j] + b_soft[j]) / TEMP ), j < 11
__global__ void __launch_bounds__(256) k_sample_op(
    const float* __restrict__ h_f, const float* __restrict__ Wsoft,
    const float* __restrict__ bsoft, NasState* st, float* out, int slot) {
    __shared__ float red[256];
    __shared__ float lg[NOPS];
    int tid = threadIdx.x;
    for (int j = 0; j < NOPS; ++j) {
        float p = 0.f;
        const float* w = Wsoft + (long)j * HID;
        for (int k = tid; k < HID; k += 256) p += h_f[k] * w[k];
        red[tid] = p; __syncthreads();
        for (int s = 128; s > 0; s >>= 1) {
            if (tid < s) red[tid] += red[tid + s];
            __syncthreads();
        }
        if (tid == 0) lg[j] = 1.0f * tanhf((red[0] + bsoft[j]) / TEMPR);
        __syncthreads();
    }
    if (tid == 0) sample_common(lg, NOPS, st, out, slot, false);
}

// ---------------------------------------------------------------------------
// Host driver
// ---------------------------------------------------------------------------
extern "C" void kernel_launch(void* const* d_in, const int* in_sizes, int n_in,
                              void* d_out, int out_size, void* d_ws, size_t ws_size,
                              hipStream_t stream) {
    const float* emb   = (const float*)d_in[0];
    const float* Wemb  = (const float*)d_in[1];
    const float* Whid  = (const float*)d_in[2];
    const float* Wv    = (const float*)d_in[3];
    const float* Wsoft = (const float*)d_in[4];
    const float* bsoft = (const float*)d_in[5];
    const float* Wih   = (const float*)d_in[6];
    const float* Whh   = (const float*)d_in[7];
    const float* bih   = (const float*)d_in[8];
    const float* bhh   = (const float*)d_in[9];
    float* out = (float*)d_out;
    char* ws = (char*)d_ws;

    size_t off = 0;
    auto alloc = [&](size_t bytes) { size_t o = off; off += (bytes + 255) & ~(size_t)255; return o; };
    size_t oWih  = alloc((size_t)GATES * HID * 2);
    size_t oWhh  = alloc((size_t)GATES * HID * 2);
    size_t oWhid = alloc((size_t)HID * HID * 2);
    size_t oEmb  = alloc((size_t)19 * HID * 2);
    size_t oP    = alloc((size_t)8 * HID * 4);
    size_t oG    = alloc((size_t)GATES * 4);
    size_t oA    = alloc((size_t)HID * 4);
    size_t oHbf  = alloc((size_t)HID * 2);
    size_t oZ    = alloc((size_t)HID * 2);
    size_t oHf   = alloc((size_t)HID * 4);
    size_t oC    = alloc((size_t)HID * 4);
    size_t oSt   = alloc(sizeof(NasState));

    __bf16* Wih_sw  = (__bf16*)(ws + oWih);
    __bf16* Whh_sw  = (__bf16*)(ws + oWhh);
    __bf16* Whid_sw = (__bf16*)(ws + oWhid);
    __bf16* emb_bf  = (__bf16*)(ws + oEmb);
    float*  P       = (float*)(ws + oP);
    float*  gates   = (float*)(ws + oG);
    float*  a_hid   = (float*)(ws + oA);
    __bf16* h_bf    = (__bf16*)(ws + oHbf);
    __bf16* zbuf    = (__bf16*)(ws + oZ);
    float*  h_f     = (float*)(ws + oHf);
    float*  c       = (float*)(ws + oC);
    NasState* st    = (NasState*)(ws + oSt);
    const int* inp_ptr = (const int*)(ws + oSt);   // inp_idx is first member

    // Setup (every call: ws is not preserved between launches)
    k_init<<<1, 256, 0, stream>>>(h_bf, zbuf, h_f, c, st);
    {
        long t1 = (long)GATES * HID;
        k_convert_swizzle<<<(unsigned)((t1 + 255) / 256), 256, 0, stream>>>(Wih, Wih_sw, t1);
        k_convert_swizzle<<<(unsigned)((t1 + 255) / 256), 256, 0, stream>>>(Whh, Whh_sw, t1);
        long t2 = (long)HID * HID;
        k_convert_swizzle<<<(unsigned)((t2 + 255) / 256), 256, 0, stream>>>(Whid, Whid_sw, t2);
        long t3 = (long)19 * HID;
        k_convert_row<<<(unsigned)((t3 + 255) / 256), 256, 0, stream>>>(emb, emb_bf, t3);
        k_precompute_P<<<64, 256, 0, stream>>>(emb, Wemb, P);
    }

    // 28 sequential controller steps
    int tn = 0, to = 0;
    for (int node = 0; node < NNODES; ++node) {
        for (int i = 0; i < 2; ++i) {
            k_gates<<<512, 128, 0, stream>>>(Wih_sw, Whh_sw, emb_bf, h_bf, zbuf,
                                             bih, bhh, inp_ptr, gates);
            k_lstm<<<8, 256, 0, stream>>>(gates, c, h_f, h_bf);
            k_ahid<<<128, 128, 0, stream>>>(Whid_sw, h_bf, zbuf, a_hid);
            k_sample_node<<<1, 256, 0, stream>>>(P, a_hid, Wv, node + 2, st, out, tn);
            ++tn;
        }
        for (int i = 0; i < 2; ++i) {
            k_gates<<<512, 128, 0, stream>>>(Wih_sw, Whh_sw, emb_bf, h_bf, zbuf,
                                             bih, bhh, inp_ptr, gates);
            k_lstm<<<8, 256, 0, stream>>>(gates, c, h_f, h_bf);
            k_sample_op<<<1, 256, 0, stream>>>(h_f, Wsoft, bsoft, st, out, 14 + to);
            ++to;
        }
    }
    (void)in_sizes; (void)n_in; (void)out_size; (void)ws_size;
}
